// MultiHeadedSpatialAttention_13417477833559
// MI455X (gfx1250) — compile-verified, matrix operand order checked by reading, not run on hardware
//
#include <hip/hip_runtime.h>

// MI455X (gfx1250) fused local-window multi-head spatial attention.
// One kernel: x-halo -> LDS, grouped 1x1 convs via v_wmma_f32_16x16x32_f16
// (K rows 0-7 / V rows 8-15 packed in one A-matrix), per-pixel 7x7 attention
// in LDS/registers, single HBM write of y. No workspace needed.

typedef __attribute__((ext_vector_type(16))) _Float16 v16h;
typedef __attribute__((ext_vector_type(8)))  float    v8f;

#define HDIM  128
#define TILE  16
#define HALO  22            // TILE + 2*PAD (PAD=3)
#define NHALO (HALO*HALO)   // 484
#define MID   8
#define KK    49

__global__ __launch_bounds__(256)
void mhsa_local_attn_kernel(const float* __restrict__ x,
                            const float* __restrict__ Wq, const float* __restrict__ bq,
                            const float* __restrict__ Wk, const float* __restrict__ bk,
                            const float* __restrict__ Wv, const float* __restrict__ bv,
                            const float* __restrict__ pos,
                            float* __restrict__ out)
{
    __shared__ _Float16 Xh[MID][HALO][HALO];   // raw x halo tile, f16 (WMMA B feed)
    __shared__ float    Kh[MID][HALO][HALO];   // K feature halo (zeroed at OOB)
    __shared__ float    Vh[MID][HALO][HALO];   // V feature halo (zeroed at OOB)
    __shared__ float    Qt[MID][TILE][TILE];   // Q features, interior tile
    __shared__ float    posS[KK];

    const int tid  = threadIdx.x;
    const int lane = tid & 31;
    const int wv   = tid >> 5;                 // 8 wave32s per block
    const int g    = blockIdx.y;
    const int b    = blockIdx.z;
    const int h0   = (blockIdx.x >> 3) * TILE;
    const int w0   = (blockIdx.x & 7)  * TILE;

    const float* xg = x + (size_t)(b*64 + g*MID) * (HDIM*HDIM);

    // ---- stage 0: x halo -> LDS as f16, zero-fill outside the image ----
    for (int idx = tid; idx < MID*NHALO; idx += 256) {
        int c   = idx / NHALO;
        int rem = idx - c*NHALO;
        int hy  = rem / HALO;
        int hx  = rem - hy*HALO;
        int gy  = h0 - 3 + hy;
        int gx  = w0 - 3 + hx;
        float v = 0.f;
        if ((unsigned)gy < (unsigned)HDIM && (unsigned)gx < (unsigned)HDIM)
            v = xg[c*HDIM*HDIM + gy*HDIM + gx];
        Xh[c][hy][hx] = (_Float16)v;
    }
    if (tid < KK) posS[tid] = pos[tid];
    __syncthreads();

    // ---- WMMA operands that are loop-invariant per wave ----
    // A (16x32 f16): rows 0-7 = Wk[g], rows 8-15 = Wv[g]; contraction K = 8
    // input channels zero-padded to 32. ISA layout: lane m (<16) holds row m,
    // K=0..7 in elements 0..7; lanes 16-31 hold K=8..15/24..31 -> all zero pad.
    v16h aKV = {};
    v16h aQ  = {};
    if (lane < 16) {
        const float* wr = (lane < 8) ? (Wk + (g*MID + lane)*MID)
                                     : (Wv + (g*MID + (lane-8))*MID);
#pragma unroll
        for (int i = 0; i < MID; ++i) aKV[i] = (_Float16)wr[i];
        if (lane < 8) {
            const float* wq = Wq + (g*MID + lane)*MID;
#pragma unroll
            for (int i = 0; i < MID; ++i) aQ[i] = (_Float16)wq[i];
        }
    }
    // C (16x16 f32): bias broadcast across the pixel (N) dimension.
    v8f cKV, cQ;
#pragma unroll
    for (int r = 0; r < 8; ++r) {
        int m  = r + ((lane >= 16) ? 8 : 0);   // D rows: lanes<16 -> r, else r+8
        cKV[r] = (m < 8) ? bk[g*MID + m] : bv[g*MID + (m-8)];
        cQ[r]  = (m < 8) ? bq[g*MID + m] : 0.f;
    }
    const int n = lane & 15;

    // ---- stage 1: K & V conv over the 22x22 halo, 16 pixels per WMMA ----
    for (int t = wv; t < (NHALO + 15)/16; t += 8) {
        v16h bm = {};                          // B: lane n holds K=0..15 (8 used)
        if (lane < 16) {
            int idx = t*16 + lane; if (idx > NHALO-1) idx = NHALO-1;
            int hy = idx / HALO, hx = idx - hy*HALO;
#pragma unroll
            for (int c = 0; c < MID; ++c) bm[c] = Xh[c][hy][hx];
        }
        v8f d = __builtin_amdgcn_wmma_f32_16x16x32_f16(false, aKV, false, bm,
                                                       (short)0, cKV, false, false);
        int idx = t*16 + n;
        if (idx < NHALO) {
            int hy = idx / HALO, hx = idx - hy*HALO;
            int gy = h0 - 3 + hy, gx = w0 - 3 + hx;
            // Reference zero-pads the conv OUTPUT; OOB halo slots must be 0.
            bool inb = (unsigned)gy < (unsigned)HDIM && (unsigned)gx < (unsigned)HDIM;
            if (lane < 16) {
#pragma unroll
                for (int r = 0; r < 8; ++r) Kh[r][hy][hx] = inb ? d[r] : 0.f;
            } else {
#pragma unroll
                for (int r = 0; r < 8; ++r) Vh[r][hy][hx] = inb ? d[r] : 0.f;
            }
        }
    }

    // ---- stage 2: Q conv over the 16x16 interior, one tile row per WMMA ----
    for (int t = wv; t < TILE; t += 8) {
        v16h bm = {};
        if (lane < 16) {
#pragma unroll
            for (int c = 0; c < MID; ++c) bm[c] = Xh[c][t+3][lane+3];
        }
        v8f d = __builtin_amdgcn_wmma_f32_16x16x32_f16(false, aQ, false, bm,
                                                       (short)0, cQ, false, false);
        if (lane < 16) {
#pragma unroll
            for (int r = 0; r < 8; ++r) Qt[r][t][lane] = d[r];
        }
    }
    __syncthreads();

    // ---- stage 3: per-pixel 7x7 attention (scores live in VGPRs) ----
    const int ty = tid >> 4;
    const int tx = tid & 15;

    float q[MID], qs = 0.f;
#pragma unroll
    for (int c = 0; c < MID; ++c) { q[c] = Qt[c][ty][tx]; qs += q[c]; }

    // A_p = dot(Q, K_shifted) + pos[p] * sum(Q)  (pos added post-padding)
    float A[KK];
    float amax = -3.4e38f;
#pragma unroll
    for (int i = 0; i < 7; ++i) {
#pragma unroll
        for (int j = 0; j < 7; ++j) {
            const int p = i*7 + j;
            float acc = posS[p] * qs;
#pragma unroll
            for (int c = 0; c < MID; ++c) acc += q[c] * Kh[c][ty+i][tx+j];
            A[p] = acc;
            amax = fmaxf(amax, acc);
        }
    }
    float s = 0.f;
#pragma unroll
    for (int p = 0; p < KK; ++p) { A[p] = __expf(A[p] - amax); s += A[p]; }
    const float inv = 1.0f / s;

    float y[MID] = {};
#pragma unroll
    for (int i = 0; i < 7; ++i) {
#pragma unroll
        for (int j = 0; j < 7; ++j) {
            const float w = A[i*7 + j];
#pragma unroll
            for (int c = 0; c < MID; ++c) y[c] += w * Vh[c][ty+i][tx+j];
        }
    }

    float* og = out + (size_t)(b*64 + g*MID) * (HDIM*HDIM) + (h0+ty)*HDIM + (w0+tx);
#pragma unroll
    for (int c = 0; c < MID; ++c) og[c*HDIM*HDIM] = y[c] * inv;
}

extern "C" void kernel_launch(void* const* d_in, const int* in_sizes, int n_in,
                              void* d_out, int out_size, void* d_ws, size_t ws_size,
                              hipStream_t stream) {
    (void)in_sizes; (void)n_in; (void)out_size; (void)d_ws; (void)ws_size;
    const float* x   = (const float*)d_in[0];
    const float* Wq  = (const float*)d_in[1];
    const float* bq  = (const float*)d_in[2];
    const float* Wk  = (const float*)d_in[3];
    const float* bk  = (const float*)d_in[4];
    const float* Wv  = (const float*)d_in[5];
    const float* bv  = (const float*)d_in[6];
    const float* pos = (const float*)d_in[7];
    float* out = (float*)d_out;

    dim3 grid(64, 8, 2);   // (8x8 pixel tiles, G=8 groups, B=2)
    dim3 block(256);       // 8 wave32s
    mhsa_local_attn_kernel<<<grid, block, 0, stream>>>(x, Wq, bq, Wk, bk, Wv, bv, pos, out);
}
